// GammaOneByOne_68401649156258
// MI455X (gfx1250) — matrix operand order repaired and checked by reading, not compiled, not run on hardware
//
#include <hip/hip_runtime.h>

typedef __attribute__((ext_vector_type(16))) _Float16 v16h;
typedef __attribute__((ext_vector_type(8)))  float    v8f;

#define NS 4096
#define NC 32
#define NA 500
#define WAVES_PER_BLOCK 8

// One wave32 per sample. Per 16-asset tile:
//   A (16x32 f16)  = W broadcast across the 16 rows   (built once)
//   B (32x16 f16)  = x[s, 0:32, a0:a0+16] tile         (16 coalesced b32 loads)
//   D = A*B  ->  every column n of D equals y[s, a0+n] (full K=32 contraction
//                in ONE v_wmma_f32_16x16x32_f16)
// Then acc += |y + b| (masked for assets >= 500), butterfly-reduce, store mean.
__global__ __launch_bounds__(32 * WAVES_PER_BLOCK)
void gamma_oneby_kernel(const float* __restrict__ x,
                        const float* __restrict__ W,
                        const float* __restrict__ b,
                        float* __restrict__ out) {
  const int lane = threadIdx.x & 31;
  const int wave = threadIdx.x >> 5;
  const int s    = blockIdx.x * WAVES_PER_BLOCK + wave;
  if (s >= NS) return;

  const int half = lane >> 4;   // which 16-lane half of the wave
  const int n    = lane & 15;   // column (asset) index within the tile

  // ---- A operand: W[k] replicated across all 16 rows ------------------
  // A k order matches B: lanes 0-15 hold K=0..15, lanes 16-31 hold K=16..31;
  // element e -> K=koffA+e  ==>  wi = koffA + e.
  const int koffA = half * 16;
  v16h amat;
#pragma unroll
  for (int e = 0; e < 16; ++e) {
    const int wi = koffA + e;
    amat[e] = (_Float16)W[wi];
  }
  const float bias = b[0];

  // ---- B operand channel range for this lane half ---------------------
  // B k order used here (matches A): VGPR j: lanes 0-15 = K{2j,2j+1}, lanes 16-31 = K{16+2j,...}
  const int kB = half * 16;
  const float* __restrict__ xs = x + (size_t)s * (NC * NA);

  float acc = 0.0f;

#pragma unroll 2
  for (int tile = 0; tile < 32; ++tile) {
    const int a0    = tile * 16;
    const int asset = a0 + n;
    const int col   = (asset < NA) ? asset : (NA - 1);   // clamp: stay in bounds
    const float* __restrict__ p = xs + col;

    // 16 independent coalesced b32 loads (each lane-half reads a contiguous
    // 64B run of 16 assets from one channel row), converted to f16 pairs.
    v16h bmat;
#pragma unroll
    for (int e = 0; e < 16; ++e) {
      bmat[e] = (_Float16)p[(size_t)(kB + e) * NA];
    }

    v8f c = {};
    c = __builtin_amdgcn_wmma_f32_16x16x32_f16(
        /*neg_a=*/false, amat, /*neg_b=*/false, bmat,
        /*c_mod=*/(short)0, c, /*reuse_a=*/false, /*reuse_b=*/false);

    // Every row of D equals y[s, a0+n]; c[0] is row 8*half of this lane's column.
    const float y = c[0] + bias;
    acc += (asset < NA) ? __builtin_fabsf(y) : 0.0f;   // select, EXEC uniform
  }

  // Butterfly sum over all 32 lanes. Lanes 16-31 hold exact duplicates of
  // lanes 0-15, so the full-wave sum is 2 * (sum over the 16 asset columns).
#pragma unroll
  for (int off = 16; off > 0; off >>= 1)
    acc += __shfl_xor(acc, off, 32);

  if (lane == 0)
    out[s] = acc * (1.0f / (2.0f * NA)) + 0.01f;   // /2 duplicate, /NA mean
}

extern "C" void kernel_launch(void* const* d_in, const int* in_sizes, int n_in,
                              void* d_out, int out_size, void* d_ws, size_t ws_size,
                              hipStream_t stream) {
  (void)in_sizes; (void)n_in; (void)out_size; (void)d_ws; (void)ws_size;
  const float* x = (const float*)d_in[0];   // (4096, 32, 500) f32
  const float* W = (const float*)d_in[1];   // (1, 32) f32
  const float* b = (const float*)d_in[2];   // (1,) f32
  float* out = (float*)d_out;               // (4096,) f32

  const int blocks = NS / WAVES_PER_BLOCK;  // 512 blocks x 256 threads (8 waves)
  gamma_oneby_kernel<<<blocks, 32 * WAVES_PER_BLOCK, 0, stream>>>(x, W, b, out);
}
